// HierarchicalGraphAttention_3204045603786
// MI455X (gfx1250) — matrix operand order repaired and checked by reading, not compiled, not run on hardware
//
#include <hip/hip_runtime.h>
#include <hip/hip_bf16.h>

typedef __attribute__((ext_vector_type(16))) __bf16 v16bf;
typedef __attribute__((ext_vector_type(8)))  __bf16 v8bf;
typedef __attribute__((ext_vector_type(8)))  float  v8f;

#define EDIM 256
#define NHEAD 8
#define HD 32
#define BATCH 2
#define GH 40
#define HWN 1600
#define PSZ 10
#define PP 100
#define NG 128                       // padded global node count (multiple of 32)
#define SCALE_QK 0.17677669529663687f // 1/sqrt(32)

// ---------------- WMMA fragment helpers (contiguous k order, same in A and B) -------
// A-frag (16x32 bf16): lane L holds row M=L%16, K = kb..kb+15 (contig),
// kb = (L/16)*16.  Two 16B loads.
__device__ inline v16bf frag_a(const __bf16* rowp, int lane) {
  const int kb = (lane >> 4) << 4;
  v8bf lo = *(const v8bf*)(rowp + kb);
  v8bf hi = *(const v8bf*)(rowp + kb + 8);
  return __builtin_shufflevector(lo, hi, 0,1,2,3,4,5,6,7,8,9,10,11,12,13,14,15);
}
// B-frag (32x16 bf16): lane L holds col N=L%16, K = (L/16)*16 + 0..15 (contig).
__device__ inline v16bf frag_b(const __bf16* p) { return *(const v16bf*)p; }

__device__ inline v8f wmma_bf16(v16bf a, v16bf b, v8f c) {
  return __builtin_amdgcn_wmma_f32_16x16x32_bf16(false, a, false, b, (short)0, c,
                                                 false, false);
}

// ---------------- prep: weights f32 -> bf16 ----------------------------------
__global__ void k_prep_w(const float* __restrict__ gatW,
                         const float* __restrict__ crossW, __bf16* __restrict__ wbf) {
  int i = blockIdx.x * blockDim.x + threadIdx.x;
  if (i >= 12 * EDIM * EDIM) return;
  int m = i >> 16, off = i & 0xFFFF;   // E*E == 65536
  float v = (m < 8) ? gatW[(size_t)m * 65536 + off] : crossW[(size_t)(m - 8) * 65536 + off];
  wbf[i] = (__bf16)v;
}

// ---------------- prep: global pooled nodes + local transposed nodes ---------
__global__ void k_prep_nodes(const float* __restrict__ x, float* gnF, __bf16* gnB,
                             float* lnF, __bf16* lnB) {
  int i = blockIdx.x * blockDim.x + threadIdx.x;
  const int totG = BATCH * NG * EDIM;
  const int totL = BATCH * HWN * EDIM;
  if (i < totG) {
    int e = i & (EDIM - 1);
    int n = (i >> 8) % NG;
    int b = i / (EDIM * NG);
    float acc = 0.f;
    if (n < PP) {
      int py = n / PSZ, px = n % PSZ;
      size_t base = ((size_t)(b * EDIM + e) * GH + py * 4) * GH + px * 4;
      #pragma unroll
      for (int dy = 0; dy < 4; ++dy)
        #pragma unroll
        for (int dx = 0; dx < 4; ++dx) acc += x[base + dy * GH + dx];
      acc *= (1.f / 16.f);
    }
    gnF[i] = acc; gnB[i] = (__bf16)acc;
  } else if (i < totG + totL) {
    int j = i - totG;
    int e = j & (EDIM - 1);
    int n = (j >> 8) % HWN;
    int b = j / (EDIM * HWN);
    float v = x[(size_t)(b * EDIM + e) * HWN + n];
    lnF[j] = v; lnB[j] = (__bf16)v;
  }
}

// ---------------- V transpose: [B][Npad][E] -> [B][E][Npad] ------------------
__global__ void k_transpose(const __bf16* __restrict__ vin, __bf16* __restrict__ vout,
                            int Npad) {
  int i = blockIdx.x * blockDim.x + threadIdx.x;
  if (i >= BATCH * Npad * EDIM) return;
  int n = i % Npad, e = (i / Npad) % EDIM, b = i / (Npad * EDIM);
  vout[i] = vin[(size_t)(b * Npad + n) * EDIM + e];
}

// ---------------- generic bf16 WMMA GEMM: out = A @ W^T + bias ---------------
// A:[R,256] row-major bf16, W:[256,256] row-major (out x in) bf16.
// Each wave computes a 16x64 tile: one A-frag reused across 4 B-frags per
// K-step (4 accumulators) to cut load-issue pressure 4x on the A side.
__global__ void k_gemm(const __bf16* __restrict__ A, const __bf16* __restrict__ W,
                       const float* __restrict__ bias, float* outF, __bf16* outB, int R) {
  int lane = threadIdx.x & 31;
  int wid = blockIdx.x * (blockDim.x >> 5) + (threadIdx.x >> 5);
  int cg = wid & 3, rt = wid >> 2;     // 4 column groups of 64
  if (rt * 16 >= R) return;
  const __bf16* arow  = A + (size_t)(rt * 16 + (lane & 15)) * EDIM;
  const __bf16* wbase = W + (size_t)(cg * 64 + (lane & 15)) * EDIM + ((lane >> 4) << 4);
  v8f acc[4] = {};
  #pragma unroll
  for (int k0 = 0; k0 < EDIM; k0 += 32) {
    v16bf a = frag_a(arow + k0, lane);
    #pragma unroll
    for (int c = 0; c < 4; ++c)
      acc[c] = wmma_bf16(a, frag_b(wbase + (size_t)c * 16 * EDIM + k0), acc[c]);
  }
  #pragma unroll
  for (int c = 0; c < 4; ++c) {
    int col = cg * 64 + c * 16 + (lane & 15);
    float bv = bias ? bias[col] : 0.f;
    #pragma unroll
    for (int r = 0; r < 8; ++r) {
      int row = rt * 16 + r + ((lane >> 4) << 3);
      float vv = acc[c][r] + bv;
      if (outF) outF[(size_t)row * EDIM + col] = vv;
      if (outB) outB[(size_t)row * EDIM + col] = (__bf16)vv;
    }
  }
}

// ---------------- fused GAT attention (softmax over BATCH axis) --------------
// q,k:[B][Npad][E] bf16; vT:[B][E][Npad] bf16; out o:[B][Npad][E] bf16.
// One wave per (i-tile, head); streams j in 32-chunks; softmax over b is
// elementwise in (i,j) so no cross-lane reduction is needed.
__global__ void k_gat_attn(const __bf16* __restrict__ q, const __bf16* __restrict__ k,
                           const __bf16* __restrict__ vT, const float* __restrict__ rel,
                           __bf16* __restrict__ outO, int N, int Npad, int isLocal) {
  __shared__ alignas(32) __bf16 alds[4][2][16][32];
  int w = threadIdx.x >> 5, lane = threadIdx.x & 31;
  int wid = blockIdx.x * (blockDim.x >> 5) + w;
  int itiles = Npad >> 4;
  if (wid >= itiles * NHEAD) return;
  int it = wid % itiles, h = wid / itiles;
  int i0 = it << 4, ln15 = lane & 15, lhi = lane >> 4;
  float relh = rel[h];
  v16bf aq0 = frag_a(q + (size_t)(0 * Npad + i0 + ln15) * EDIM + h * HD, lane);
  v16bf aq1 = frag_a(q + (size_t)(1 * Npad + i0 + ln15) * EDIM + h * HD, lane);
  v8f o0[2] = {}, o1[2] = {};
  for (int j0 = 0; j0 < Npad; j0 += 32) {
    v8f s0[2], s1[2];
    #pragma unroll
    for (int t = 0; t < 2; ++t) {
      const __bf16* k0p = k + (size_t)(0 * Npad + j0 + 16 * t + ln15) * EDIM + h * HD + (lhi << 4);
      const __bf16* k1p = k + (size_t)(1 * Npad + j0 + 16 * t + ln15) * EDIM + h * HD + (lhi << 4);
      v8f z = {};
      s0[t] = wmma_bf16(aq0, frag_b(k0p), z);
      s1[t] = wmma_bf16(aq1, frag_b(k1p), z);
    }
    #pragma unroll
    for (int t = 0; t < 2; ++t)
      #pragma unroll
      for (int r = 0; r < 8; ++r) {
        int i = i0 + r + (lhi << 3);
        int j = j0 + 16 * t + ln15;
        bool valid = (i < N) && (j < N);
        float bias = 0.f;
        if (isLocal) {
          int yi = i / GH, xi = i - yi * GH, yj = j / GH, xj = j - yj * GH;
          int dy = yi - yj, dx = xi - xj;
          if (dy < 0) dy = -dy; if (dx < 0) dx = -dx;
          if (dy + dx == 1) bias = relh;          // 4-neighbor grid edge
        } else if (valid) bias = relh;            // dense ones edges
        float a0v = s0[t][r] * SCALE_QK + bias;
        float a1v = s1[t][r] * SCALE_QK + bias;
        float m = fmaxf(a0v, a1v);
        float e0 = __expf(a0v - m), e1 = __expf(a1v - m);
        float inv = 1.f / (e0 + e1);              // softmax over batch (size 2)
        int M = r + (lhi << 3);
        alds[w][0][M][16 * t + ln15] = (__bf16)(valid ? e0 * inv : 0.f);
        alds[w][1][M][16 * t + ln15] = (__bf16)(valid ? e1 * inv : 0.f);
      }
    asm volatile("s_wait_dscnt 0" ::: "memory");
    v16bf a0f = frag_a(&alds[w][0][ln15][0], lane);
    v16bf a1f = frag_a(&alds[w][1][ln15][0], lane);
    #pragma unroll
    for (int t2 = 0; t2 < 2; ++t2) {
      const __bf16* v0p = vT + (size_t)(0 * EDIM + h * HD + 16 * t2 + ln15) * Npad + j0 + (lhi << 4);
      const __bf16* v1p = vT + (size_t)(1 * EDIM + h * HD + 16 * t2 + ln15) * Npad + j0 + (lhi << 4);
      o0[t2] = wmma_bf16(a0f, frag_b(v0p), o0[t2]);
      o1[t2] = wmma_bf16(a1f, frag_b(v1p), o1[t2]);
    }
  }
  #pragma unroll
  for (int t2 = 0; t2 < 2; ++t2)
    #pragma unroll
    for (int r = 0; r < 8; ++r) {
      int i = i0 + r + (lhi << 3);
      if (i < N) {
        outO[(size_t)(0 * Npad + i) * EDIM + h * HD + 16 * t2 + ln15] = (__bf16)o0[t2][r];
        outO[(size_t)(1 * Npad + i) * EDIM + h * HD + 16 * t2 + ln15] = (__bf16)o1[t2][r];
      }
    }
}

// ---------------- fused cross attention over 100 unique global KV ------------
// (x4 nearest-upsample duplicates each KV 16x; duplicates cancel in softmax)
__global__ void k_cross_attn(const __bf16* __restrict__ q, const __bf16* __restrict__ k,
                             const __bf16* __restrict__ vT, __bf16* __restrict__ outO) {
  __shared__ alignas(32) __bf16 alds[4][16][NG];
  int w = threadIdx.x >> 5, lane = threadIdx.x & 31;
  int wid = blockIdx.x * (blockDim.x >> 5) + w;
  const int itiles = HWN >> 4; // 100
  if (wid >= itiles * NHEAD * BATCH) return;
  int it = wid % itiles, hb = wid / itiles, h = hb % NHEAD, b = hb / NHEAD;
  int i0 = it << 4, ln15 = lane & 15, lhi = lane >> 4;
  v16bf aq = frag_a(q + (size_t)(b * HWN + i0 + ln15) * EDIM + h * HD, lane);
  v8f s[8];
  #pragma unroll
  for (int t = 0; t < 8; ++t) {
    const __bf16* kp = k + (size_t)(b * NG + 16 * t + ln15) * EDIM + h * HD + (lhi << 4);
    v8f z = {};
    s[t] = wmma_bf16(aq, frag_b(kp), z);
  }
  // masked softmax over j in [0,PP); rows 0..7 live in lanes 0-15, 8..15 in 16-31
  #pragma unroll
  for (int r = 0; r < 8; ++r) {
    float pm = -1e30f;
    #pragma unroll
    for (int t = 0; t < 8; ++t) {
      float sv = s[t][r] * SCALE_QK;
      s[t][r] = sv;
      if (16 * t + ln15 < PP) pm = fmaxf(pm, sv);
    }
    #pragma unroll
    for (int m = 1; m < 16; m <<= 1) pm = fmaxf(pm, __shfl_xor(pm, m, 32));
    float ds = 0.f;
    #pragma unroll
    for (int t = 0; t < 8; ++t) {
      float ev = (16 * t + ln15 < PP) ? __expf(s[t][r] - pm) : 0.f;
      s[t][r] = ev; ds += ev;
    }
    #pragma unroll
    for (int m = 1; m < 16; m <<= 1) ds += __shfl_xor(ds, m, 32);
    float inv = 1.f / ds;
    #pragma unroll
    for (int t = 0; t < 8; ++t) s[t][r] *= inv;
  }
  #pragma unroll
  for (int t = 0; t < 8; ++t)
    #pragma unroll
    for (int r = 0; r < 8; ++r)
      alds[w][r + (lhi << 3)][16 * t + ln15] = (__bf16)s[t][r];
  asm volatile("s_wait_dscnt 0" ::: "memory");
  v8f o[2] = {};
  #pragma unroll
  for (int jc = 0; jc < 4; ++jc) {
    v16bf af = frag_a(&alds[w][ln15][jc * 32], lane);
    #pragma unroll
    for (int t2 = 0; t2 < 2; ++t2) {
      const __bf16* vp = vT + (size_t)(b * EDIM + h * HD + 16 * t2 + ln15) * NG + jc * 32 + (lhi << 4);
      o[t2] = wmma_bf16(af, frag_b(vp), o[t2]);
    }
  }
  #pragma unroll
  for (int t2 = 0; t2 < 2; ++t2)
    #pragma unroll
    for (int r = 0; r < 8; ++r) {
      int i = i0 + r + (lhi << 3);
      outO[(size_t)(b * HWN + i) * EDIM + h * HD + 16 * t2 + ln15] = (__bf16)o[t2][r];
    }
}

// ---------------- residual + LayerNorm (one wave per row) --------------------
__global__ void k_ln(const float* __restrict__ xin, const float* __restrict__ resid,
                     const float* __restrict__ g, const float* __restrict__ bta,
                     float* outF, __bf16* outB, float* outT, int N, int Npad) {
  int lane = threadIdx.x & 31;
  int wid = blockIdx.x * (blockDim.x >> 5) + (threadIdx.x >> 5);
  if (wid >= BATCH * N) return;
  int n = wid % N, b = wid / N;
  size_t row = (size_t)(b * Npad + n) * EDIM;
  float v[8], s = 0.f;
  #pragma unroll
  for (int t = 0; t < 8; ++t) { int e = t * 32 + lane; v[t] = xin[row + e] + resid[row + e]; s += v[t]; }
  #pragma unroll
  for (int m = 1; m < 32; m <<= 1) s += __shfl_xor(s, m, 32);
  float mean = s * (1.f / EDIM), vs = 0.f;
  #pragma unroll
  for (int t = 0; t < 8; ++t) { float d = v[t] - mean; vs += d * d; }
  #pragma unroll
  for (int m = 1; m < 32; m <<= 1) vs += __shfl_xor(vs, m, 32);
  float rstd = rsqrtf(vs * (1.f / EDIM) + 1e-5f);
  #pragma unroll
  for (int t = 0; t < 8; ++t) {
    int e = t * 32 + lane;
    float y = (v[t] - mean) * rstd * g[e] + bta[e];
    if (outF) outF[row + e] = y;
    if (outB) outB[row + e] = (__bf16)y;
    if (outT) outT[(size_t)(b * EDIM + e) * HWN + n] = y;  // -> [B,C,H,W]
  }
}

// ---------------- host orchestration -----------------------------------------
extern "C" void kernel_launch(void* const* d_in, const int* in_sizes, int n_in,
                              void* d_out, int out_size, void* d_ws, size_t ws_size,
                              hipStream_t stream) {
  (void)in_sizes; (void)n_in; (void)out_size; (void)ws_size;
  const float* x      = (const float*)d_in[0];
  const float* gatW   = (const float*)d_in[1];
  const float* gatB   = (const float*)d_in[2];
  const float* gatRel = (const float*)d_in[3];
  const float* crossW = (const float*)d_in[4];
  const float* crossB = (const float*)d_in[5];
  const float* lnG    = (const float*)d_in[6];
  const float* lnBi   = (const float*)d_in[7];

  char* p = (char*)d_ws;
  auto carve = [&](size_t bytes) -> void* {
    void* r = (void*)p; p += (bytes + 255) & ~(size_t)255; return r;
  };
  const size_t GROW = (size_t)BATCH * NG * EDIM;   // global rows*E (padded)
  const size_t LROW = (size_t)BATCH * HWN * EDIM;  // local rows*E

  __bf16* wbf   = (__bf16*)carve(12 * EDIM * EDIM * 2);
  float*  gnF   = (float*)carve(GROW * 4);  __bf16* gnB   = (__bf16*)carve(GROW * 2);
  float*  lnF   = (float*)carve(LROW * 4);  __bf16* lnB   = (__bf16*)carve(LROW * 2);
  __bf16* qg  = (__bf16*)carve(GROW * 2); __bf16* kg  = (__bf16*)carve(GROW * 2);
  __bf16* vg  = (__bf16*)carve(GROW * 2); __bf16* vgT = (__bf16*)carve(GROW * 2);
  __bf16* og  = (__bf16*)carve(GROW * 2);
  float*  gprojF = (float*)carve(GROW * 4);
  float*  goutF  = (float*)carve(GROW * 4); __bf16* goutB = (__bf16*)carve(GROW * 2);
  __bf16* ql  = (__bf16*)carve(LROW * 2); __bf16* kl  = (__bf16*)carve(LROW * 2);
  __bf16* vl  = (__bf16*)carve(LROW * 2); __bf16* vlT = (__bf16*)carve(LROW * 2);
  __bf16* ol  = (__bf16*)carve(LROW * 2);
  float*  lprojF = (float*)carve(LROW * 4);
  float*  loutF  = (float*)carve(LROW * 4); __bf16* loutB = (__bf16*)carve(LROW * 2);
  __bf16* qc  = (__bf16*)carve(LROW * 2);
  __bf16* kc  = (__bf16*)carve(GROW * 2); __bf16* vc  = (__bf16*)carve(GROW * 2);
  __bf16* vcT = (__bf16*)carve(GROW * 2); __bf16* oc  = (__bf16*)carve(LROW * 2);
  float*  cprojF = (float*)carve(LROW * 4);

  const int RG = BATCH * NG;   // 256 padded global rows
  const int RL = BATCH * HWN;  // 3200 local rows
  auto gemm = [&](const __bf16* A, int widx, const float* bias, float* oF, __bf16* oB, int R) {
    int tiles = (R / 16) * 4;  // 16x64 tile per wave, 4 column groups
    k_gemm<<<tiles / 4, 128, 0, stream>>>(A, wbf + (size_t)widx * EDIM * EDIM, bias, oF, oB, R);
  };

  k_prep_w<<<(12 * EDIM * EDIM) / 256, 256, 0, stream>>>(gatW, crossW, wbf);
  k_prep_nodes<<<(int)((GROW + LROW) / 256), 256, 0, stream>>>(x, gnF, gnB, lnF, lnB);

  // ---- global GAT ----
  gemm(gnB, 0, gatB + 0 * EDIM, nullptr, qg, RG);
  gemm(gnB, 1, gatB + 1 * EDIM, nullptr, kg, RG);
  gemm(gnB, 2, gatB + 2 * EDIM, nullptr, vg, RG);
  k_transpose<<<(int)(GROW / 256), 256, 0, stream>>>(vg, vgT, NG);
  k_gat_attn<<<(NG / 16) * NHEAD / 4, 128, 0, stream>>>(qg, kg, vgT, gatRel, og, PP, NG, 0);
  gemm(og, 3, gatB + 3 * EDIM, gprojF, nullptr, RG);
  k_ln<<<(BATCH * PP) / 4, 128, 0, stream>>>(gprojF, gnF, lnG, lnBi, goutF, goutB, nullptr, PP, NG);

  // ---- local GAT ----
  gemm(lnB, 4, gatB + 4 * EDIM, nullptr, ql, RL);
  gemm(lnB, 5, gatB + 5 * EDIM, nullptr, kl, RL);
  gemm(lnB, 6, gatB + 6 * EDIM, nullptr, vl, RL);
  k_transpose<<<(int)(LROW / 256), 256, 0, stream>>>(vl, vlT, HWN);
  k_gat_attn<<<(HWN / 16) * NHEAD / 4, 128, 0, stream>>>(ql, kl, vlT, gatRel + NHEAD, ol, HWN, HWN, 1);
  gemm(ol, 7, gatB + 7 * EDIM, lprojF, nullptr, RL);
  k_ln<<<(BATCH * HWN) / 4, 128, 0, stream>>>(lprojF, lnF, lnG + EDIM, lnBi + EDIM, loutF, loutB, nullptr, HWN, HWN);

  // ---- cross attention (unique global KV) ----
  gemm(loutB, 8,  crossB + 0 * EDIM, nullptr, qc, RL);
  gemm(goutB, 9,  crossB + 1 * EDIM, nullptr, kc, RG);
  gemm(goutB, 10, crossB + 2 * EDIM, nullptr, vc, RG);
  k_transpose<<<(int)(GROW / 256), 256, 0, stream>>>(vc, vcT, NG);
  k_cross_attn<<<(HWN / 16) * NHEAD * BATCH / 4, 128, 0, stream>>>(qc, kc, vcT, oc);
  gemm(oc, 11, crossB + 3 * EDIM, cprojF, nullptr, RL);
  k_ln<<<(BATCH * HWN) / 4, 128, 0, stream>>>(cprojF, loutF, lnG + 2 * EDIM, lnBi + 2 * EDIM,
                                              nullptr, nullptr, (float*)d_out, HWN, HWN);
}
